// PhysicsEngine_45664092291393
// MI455X (gfx1250) — compile-verified, matrix operand order checked by reading, not run on hardware
//
#include <hip/hip_runtime.h>

// ---------------------------------------------------------------------------
// PhysicsEngine radius-graph message passing for MI455X (gfx1250, wave32).
// Per edge: rel = y[j]-x[i]; h = gelu(rel@W1+b1); h = gelu(h@W2+b2);
//           k = h@W3+b3; msg = k*f_y[j]; segment-mean over x_index.
// Middle layer (128x256, 97% of FLOPs) runs on v_wmma_f32_16x16x32_f16,
// with W2 pre-packed to f16 B-fragment order and staged in LDS per block.
// ---------------------------------------------------------------------------

typedef __attribute__((ext_vector_type(16))) _Float16 v16h;
typedef __attribute__((ext_vector_type(8)))  float    v8f;

#define H1   128
#define H2   256
#define DIMS 3
#define EDGES_PER_WAVE  16
#define WAVES_PER_BLOCK 8
#define EDGES_PER_BLOCK (EDGES_PER_WAVE * WAVES_PER_BLOCK)  // 128

union AFrag { v16h v; _Float16 h[16]; };
union CFrag { v8f  v; float    f[8];  };

// Branchless tanh-GELU. Prefer the gfx1250 hardware transcendental
// v_tanh_f32; otherwise use the sigmoid identity
//   0.5*x*(1+tanh(u)) == x * sigmoid(2u)
// with the native (branch-free) exp path.
__device__ __forceinline__ float gelu_tanh(float x) {
    const float kAlpha = 0.7978845608028654f;   // sqrt(2/pi)
    const float kBeta  = 0.044715f;
    float u = kAlpha * (x + kBeta * x * x * x);
#if __has_builtin(__builtin_amdgcn_tanhf)
    return 0.5f * x * (1.0f + __builtin_amdgcn_tanhf(u));
#else
    float e = __expf(-2.0f * u);
    return __fdividef(x, 1.0f + e);
#endif
}

// ---------------------------------------------------------------------------
// Pack W2 (128x256 f32, row-major [k][n]) into f16 B-fragment order.
// 16-bit B 32x16 fragment layout (mirror of documented A layout):
//   lane l holds N = l%16;  half t holds K = 16*(t/8) + 8*(l/16) + (t%8)
// fragment index = (n_tile*4 + k_chunk); 512 halves per fragment.
// ---------------------------------------------------------------------------
__global__ __launch_bounds__(256)
void pack_w2_kernel(const float* __restrict__ W2, _Float16* __restrict__ out) {
    int idx = blockIdx.x * 256 + threadIdx.x;       // 0 .. 128*256-1
    if (idx >= H1 * H2) return;
    int k  = idx >> 8;          // 0..127
    int n  = idx & 255;         // 0..255
    int c  = k >> 5;            // K chunk of 32
    int kk = k & 31;
    int nt = n >> 4;            // N tile of 16
    int nn = n & 15;
    int g  = kk >> 4;           // which 16-group inside the chunk
    int hi = ((kk & 15) >= 8) ? 1 : 0;
    int t  = g * 8 + (kk & 7);
    int l  = nn + 16 * hi;
    out[(size_t)((nt * 4 + c) * 512 + l * 16 + t)] = (_Float16)W2[idx];
}

__global__ __launch_bounds__(256)
void zero_kernel(float* __restrict__ p, int n) {
    int i = blockIdx.x * 256 + threadIdx.x;
    if (i < n) p[i] = 0.0f;
}

// ---------------------------------------------------------------------------
// Main edge kernel: one wave processes 16 edges end-to-end.
// ---------------------------------------------------------------------------
__global__ __launch_bounds__(256)
void edge_mlp_kernel(const float* __restrict__ y,
                     const float* __restrict__ x,
                     const float* __restrict__ fy,
                     const int*   __restrict__ nbr,
                     const int*   __restrict__ xidx,
                     const float* __restrict__ W1,
                     const float* __restrict__ b1,
                     const _Float16* __restrict__ w2pk,
                     const float* __restrict__ b2,
                     const float* __restrict__ W3,
                     const float* __restrict__ b3,
                     float* __restrict__ num,
                     float* __restrict__ cnt,
                     int E)
{
    __shared__ __align__(16) _Float16 sW2[H1 * H2];   // 64 KB, fragment order
    __shared__ float sW1[DIMS * H1];
    __shared__ float sB1[H1];
    __shared__ float sB2[H2];
    __shared__ float sW3[H2 * DIMS];
    __shared__ float sB3[DIMS];
    __shared__ float sRel[WAVES_PER_BLOCK][EDGES_PER_WAVE][DIMS];
    __shared__ float sFy [WAVES_PER_BLOCK][EDGES_PER_WAVE][DIMS];
    __shared__ int   sXi [WAVES_PER_BLOCK][EDGES_PER_WAVE];

    const int tid  = threadIdx.x;
    const int wave = tid >> 5;
    const int lane = tid & 31;
    const int hi   = lane >> 4;      // 0: lanes 0-15, 1: lanes 16-31
    const int nlo  = lane & 15;

    // Stage packed W2 into LDS (block-wide, read by all 8 waves).
    {
        const float4* src = (const float4*)w2pk;      // 4096 x 16B
        float4*       dst = (float4*)sW2;
        #pragma unroll
        for (int i = 0; i < 16; ++i) {
            int idx = i * 256 + tid;
            __builtin_prefetch((const void*)(src + idx + 256), 0, 0);
            dst[idx] = src[idx];
        }
    }
    // Stage small weights in LDS.
    for (int i = tid; i < DIMS * H1; i += 256) sW1[i] = W1[i];
    for (int i = tid; i < H1;        i += 256) sB1[i] = b1[i];
    for (int i = tid; i < H2;        i += 256) sB2[i] = b2[i];
    for (int i = tid; i < H2 * DIMS; i += 256) sW3[i] = W3[i];
    if (tid < DIMS) sB3[tid] = b3[tid];

    // Stage this wave's 16 edges: rel, f_y[j], x_index.
    if (lane < EDGES_PER_WAVE) {
        int e = blockIdx.x * EDGES_PER_BLOCK + wave * EDGES_PER_WAVE + lane;
        if (e < E) {
            int j = nbr[e];
            int i = xidx[e];
            sXi[wave][lane] = i;
            #pragma unroll
            for (int c = 0; c < DIMS; ++c) {
                sRel[wave][lane][c] = y[j * DIMS + c] - x[i * DIMS + c];
                sFy [wave][lane][c] = fy[j * DIMS + c];
            }
        } else {
            sXi[wave][lane] = -1;
            #pragma unroll
            for (int c = 0; c < DIMS; ++c) {
                sRel[wave][lane][c] = 0.0f;
                sFy [wave][lane][c] = 0.0f;
            }
        }
    }
    __syncthreads();

    // ---- Layer 1 (3 -> 128) on VALU, producing WMMA A-fragments in f16. ----
    // A layout (16-bit 16x32): row M = lane%16 for all lanes; half t of chunk c
    // holds K = 32c + 16*(t/8) + 8*(lane/16) + (t%8).
    const float r0 = sRel[wave][nlo][0];
    const float r1 = sRel[wave][nlo][1];
    const float r2 = sRel[wave][nlo][2];

    AFrag af[4];
    #pragma unroll
    for (int c2 = 0; c2 < 4; ++c2) {
        #pragma unroll
        for (int t = 0; t < 16; ++t) {
            int k = c2 * 32 + (t >> 3) * 16 + hi * 8 + (t & 7);
            float h = sB1[k] + r0 * sW1[0 * H1 + k]
                             + r1 * sW1[1 * H1 + k]
                             + r2 * sW1[2 * H1 + k];
            af[c2].h[t] = (_Float16)gelu_tanh(h);
        }
    }

    // ---- Layer 2 (128 -> 256) on WMMA + fused layer 3 (256 -> 3) partials. --
    // C/D layout: VGPR r, lane l -> M = r + 8*(l/16), N = l%16.
    float part[8][DIMS];
    #pragma unroll
    for (int r = 0; r < 8; ++r)
        #pragma unroll
        for (int c = 0; c < DIMS; ++c) part[r][c] = 0.0f;

    #pragma unroll
    for (int nt = 0; nt < 16; ++nt) {
        CFrag acc;
        const float bias = sB2[nt * 16 + nlo];
        #pragma unroll
        for (int r = 0; r < 8; ++r) acc.f[r] = bias;

        #pragma unroll
        for (int c2 = 0; c2 < 4; ++c2) {
            const v16h* bp = (const v16h*)(sW2 +
                ((nt * 4 + c2) * 512 + lane * 16));
            v16h bfrag = *bp;                         // 2x ds_load_b128
            acc.v = __builtin_amdgcn_wmma_f32_16x16x32_f16(
                        false, af[c2].v, false, bfrag,
                        (short)0, acc.v, false, false);
        }

        // GELU + fold into layer-3 partial sums (this lane's column of W3).
        const int col = nt * 16 + nlo;
        const float w30 = sW3[col * DIMS + 0];
        const float w31 = sW3[col * DIMS + 1];
        const float w32 = sW3[col * DIMS + 2];
        #pragma unroll
        for (int r = 0; r < 8; ++r) {
            float h2 = gelu_tanh(acc.f[r]);
            part[r][0] += h2 * w30;
            part[r][1] += h2 * w31;
            part[r][2] += h2 * w32;
        }
    }

    // ---- Reduce layer-3 partials across the 16 N-lanes of each half. -------
    #pragma unroll
    for (int off = 1; off < 16; off <<= 1) {
        #pragma unroll
        for (int r = 0; r < 8; ++r) {
            #pragma unroll
            for (int c = 0; c < DIMS; ++c)
                part[r][c] += __shfl_xor(part[r][c], off, 16);
        }
    }

    // Lanes 0 and 16 own the reduced sums for rows 0-7 / 8-15 respectively.
    if (nlo == 0) {
        #pragma unroll
        for (int r = 0; r < 8; ++r) {
            int mrow = r + hi * 8;
            int i = sXi[wave][mrow];
            if (i >= 0) {
                #pragma unroll
                for (int c = 0; c < DIMS; ++c) {
                    float msg = (part[r][c] + sB3[c]) * sFy[wave][mrow][c];
                    atomicAdd(&num[(size_t)i * DIMS + c], msg);
                }
                atomicAdd(&cnt[i], 1.0f);
            }
        }
    }
}

__global__ __launch_bounds__(256)
void finalize_kernel(const float* __restrict__ num,
                     const float* __restrict__ cnt,
                     float* __restrict__ out, int nx)
{
    int i = blockIdx.x * 256 + threadIdx.x;
    if (i >= nx) return;
    float inv = 1.0f / fmaxf(cnt[i], 1.0f);
    #pragma unroll
    for (int c = 0; c < DIMS; ++c)
        out[(size_t)i * DIMS + c] = num[(size_t)i * DIMS + c] * inv;
}

// ---------------------------------------------------------------------------
// Host launcher. Inputs in setup_inputs() dict order:
// 0 rom_ic(y) 1 fom_ic(x) 2 rom_f 3 nbr_index 4 x_index
// 5 W1 6 b1 7 W2 8 b2 9 W3 10 b3
// jnp.asarray(int64) under default JAX yields int32 on device -> const int*.
// ---------------------------------------------------------------------------
extern "C" void kernel_launch(void* const* d_in, const int* in_sizes, int n_in,
                              void* d_out, int out_size, void* d_ws, size_t ws_size,
                              hipStream_t stream) {
    const float* y   = (const float*)d_in[0];
    const float* x   = (const float*)d_in[1];
    const float* fy  = (const float*)d_in[2];
    const int*   nbr = (const int*)d_in[3];
    const int*   xid = (const int*)d_in[4];
    const float* W1  = (const float*)d_in[5];
    const float* b1  = (const float*)d_in[6];
    const float* W2  = (const float*)d_in[7];
    const float* b2  = (const float*)d_in[8];
    const float* W3  = (const float*)d_in[9];
    const float* b3  = (const float*)d_in[10];

    const int E  = in_sizes[3];
    const int NX = in_sizes[1] / DIMS;

    // Workspace layout: [0,64KB) packed f16 W2; then num (NX*3) and cnt (NX).
    _Float16* w2pk = (_Float16*)d_ws;
    float*    num  = (float*)((char*)d_ws + (size_t)(H1 * H2 * 2));
    float*    cnt  = num + (size_t)NX * DIMS;
    float*    out  = (float*)d_out;

    pack_w2_kernel<<<(H1 * H2 + 255) / 256, 256, 0, stream>>>(W2, w2pk);
    zero_kernel<<<(NX * 4 + 255) / 256, 256, 0, stream>>>(num, NX * 4);

    if (E > 0) {
        int nblk = (E + EDGES_PER_BLOCK - 1) / EDGES_PER_BLOCK;
        edge_mlp_kernel<<<nblk, 256, 0, stream>>>(
            y, x, fy, nbr, xid, W1, b1, w2pk, b2, W3, b3, num, cnt, E);
    }

    finalize_kernel<<<(NX + 255) / 256, 256, 0, stream>>>(num, cnt, out, NX);
}
